// MultiHeadAttention_70145405878378
// MI455X (gfx1250) — matrix operand order repaired and checked by reading, not compiled, not run on hardware
//
#include <hip/hip_runtime.h>

#define B_ 4
#define S_ 2048
#define H_ 16
#define D_ 64
#define E_ 1024

typedef __attribute__((ext_vector_type(16))) __bf16 v16bf;
typedef __attribute__((ext_vector_type(8)))  float  v8f;
typedef __attribute__((ext_vector_type(4)))  unsigned int v4u;

union FragU { v16bf v; v4u u[2]; };
union PackU { v4u u; __bf16 h[8]; };

// ---- WMMA fragment loads from LDS (layouts per CDNA5 ISA 7.12.2) ----
// A (16x32 bf16): lane<16: row=lane, K in [0..8)+[16..24); lane>=16: row=lane-16, K in [8..16)+[24..32)
__device__ __forceinline__ v16bf frag_a_lds(const __bf16* p, int stride) {
  const int lane = threadIdx.x & 31;
  const int row  = lane & 15;
  const int kb   = (lane >> 4) << 3;        // 0 or 8 elements
  FragU f;
  f.u[0] = *(const v4u*)(p + row * stride + kb);
  f.u[1] = *(const v4u*)(p + row * stride + kb + 16);
  return f.v;
}
// B (32x16 bf16), stored in LDS as [n][k] (each "row" = one output column, K contiguous):
// lane<16: col=lane, K in [0..8)+[16..24); lane>=16: col=lane-16, K in [8..16)+[24..32)
__device__ __forceinline__ v16bf frag_b_lds(const __bf16* p, int stride) {
  const int lane = threadIdx.x & 31;
  const int col  = lane & 15;
  const int kb   = (lane >> 4) << 3;        // 0 or 8 elements
  FragU f;
  f.u[0] = *(const v4u*)(p + col * stride + kb);
  f.u[1] = *(const v4u*)(p + col * stride + kb + 16);
  return f.v;
}

__device__ __forceinline__ v4u pack8(float4 a, float4 b) {
  PackU p;
  p.h[0] = (__bf16)a.x; p.h[1] = (__bf16)a.y; p.h[2] = (__bf16)a.z; p.h[3] = (__bf16)a.w;
  p.h[4] = (__bf16)b.x; p.h[5] = (__bf16)b.y; p.h[6] = (__bf16)b.z; p.h[7] = (__bf16)b.w;
  return p.u;
}

// ---- CDNA5 async copy: global -> LDS, tracked by ASYNCcnt ----
// GVS mode: mem_addr = SADDR(64b sgpr, uniform) + VADDR(32b vgpr byte offset).
// VDST vgpr holds per-lane LDS byte address.
__device__ __forceinline__ void async_ld_b128(unsigned lds_addr, unsigned voff,
                                              const void* sbase) {
  asm volatile("global_load_async_to_lds_b128 %0, %1, %2"
               :: "v"(lds_addr), "v"(voff), "s"(sbase)
               : "memory");
}
__device__ __forceinline__ void wait_async0() {
  asm volatile("s_wait_asynccnt 0x0" ::: "memory");
}

// =====================================================================
// Kernel 1: QKV projection GEMM (bf16 WMMA) + fused RoPE + L2norm + scale
// grid: (M/128, 3*H). blockIdx.y: which = y/16 (0=q,1=k,2=v), head = y%16
// Workgroup tile 128(M) x 64(N); 8 waves, each wave 16(M) x 64(N) = 4 accs.
// Staging keeps the register double-buffer (fp32->bf16 conversion happens
// in-flight, which the raw async DMA path cannot do).
// q,k -> bf16 (B,H,S,D); v -> bf16 transposed (B,H,D,S)
// =====================================================================
__global__ __launch_bounds__(256)
void qkv_rope_norm_kernel(const float* __restrict__ x,
                          const float* __restrict__ Wq,
                          const float* __restrict__ Wk,
                          const float* __restrict__ Wv,
                          const float* __restrict__ sqk,
                          const float* __restrict__ fcos,
                          const float* __restrict__ fsin,
                          __bf16* __restrict__ qb,
                          __bf16* __restrict__ kb,
                          __bf16* __restrict__ vtb) {
  __shared__ __align__(16) __bf16 lA[128][40];   // x tile (128 rows x 32 K), padded pitch
  __shared__ __align__(16) __bf16 lB[64][40];    // W tile  (64 cols x 32 K), padded pitch

  const int tid  = threadIdx.x;
  const int wv   = tid >> 5;
  const int lane = tid & 31;
  const int m0   = blockIdx.x * 128;
  const int which = blockIdx.y >> 4;             // 0=q 1=k 2=v
  const int h    = blockIdx.y & 15;
  const int n0   = h * 64;
  const float* W = (which == 0) ? Wq : (which == 1) ? Wk : Wv;

  const v8f vzero = {0.f, 0.f, 0.f, 0.f, 0.f, 0.f, 0.f, 0.f};
  v8f acc[4];
#pragma unroll
  for (int t = 0; t < 4; ++t) acc[t] = vzero;

  const int arow = tid >> 1, ahalf = tid & 1;    // x staging: 128 rows x 2 halves
  const int brow = tid >> 2, bq    = tid & 3;    // W staging: 64 rows x 4 quarters
  const float* aptr = x + (size_t)(m0 + arow) * E_ + ahalf * 16;
  const float* bptr = W + (size_t)(n0 + brow) * E_ + bq * 8;

  // prologue: prefetch tile 0 into registers
  float4 fa0 = ((const float4*)aptr)[0], fa1 = ((const float4*)aptr)[1];
  float4 fa2 = ((const float4*)aptr)[2], fa3 = ((const float4*)aptr)[3];
  float4 gb0 = ((const float4*)bptr)[0], gb1 = ((const float4*)bptr)[1];

  for (int kk = 0; kk < E_ / 32; ++kk) {
    __syncthreads();                    // previous compute done; LDS reusable
    *(v4u*)&lA[arow][ahalf * 16]     = pack8(fa0, fa1);
    *(v4u*)&lA[arow][ahalf * 16 + 8] = pack8(fa2, fa3);
    *(v4u*)&lB[brow][bq * 8]         = pack8(gb0, gb1);
    __syncthreads();

    if (kk + 1 < E_ / 32) {             // issue next tile's loads before compute
      const float4* sa = (const float4*)(aptr + (kk + 1) * 32);
      fa0 = sa[0]; fa1 = sa[1]; fa2 = sa[2]; fa3 = sa[3];
      const float4* sb = (const float4*)(bptr + (kk + 1) * 32);
      gb0 = sb[0]; gb1 = sb[1];
    }

    // hoist all fragment loads, then issue WMMAs back-to-back
    const v16bf a  = frag_a_lds(&lA[wv * 16][0], 40);
    const v16bf b0 = frag_b_lds(&lB[0][0],  40);
    const v16bf b1 = frag_b_lds(&lB[16][0], 40);
    const v16bf b2 = frag_b_lds(&lB[32][0], 40);
    const v16bf b3 = frag_b_lds(&lB[48][0], 40);
    acc[0] = __builtin_amdgcn_wmma_f32_16x16x32_bf16(false, a, false, b0, (short)0, acc[0], false, false);
    acc[1] = __builtin_amdgcn_wmma_f32_16x16x32_bf16(false, a, false, b1, (short)0, acc[1], false, false);
    acc[2] = __builtin_amdgcn_wmma_f32_16x16x32_bf16(false, a, false, b2, (short)0, acc[2], false, false);
    acc[3] = __builtin_amdgcn_wmma_f32_16x16x32_bf16(false, a, false, b3, (short)0, acc[3], false, false);
  }

  // ---- epilogue ----
  const int hi = lane >> 4;
  const int cb = lane & 15;
  const int mrow0 = m0 + wv * 16;
  const int bidx  = mrow0 >> 11;                 // batch (S_=2048)
  const int s0    = (mrow0 & (S_ - 1)) + hi * 8; // sequence pos of element r=0
  const int bh    = bidx * H_ + h;

  if (which < 2) {
    // RoPE (interleaved pairs; partner element lives in lane^1)
#pragma unroll
    for (int t = 0; t < 4; ++t) {
      const int d = t * 16 + cb;
      const int i = d >> 1;
      const bool even = (d & 1) == 0;
#pragma unroll
      for (int r = 0; r < 8; ++r) {
        const int pos = s0 + r;
        const float c = fcos[pos * (D_ / 2) + i];
        const float s = fsin[pos * (D_ / 2) + i];
        const float vv = acc[t][r];
        const float ov = __shfl_xor(vv, 1, 32);
        acc[t][r] = even ? (vv * c - ov * s) : (ov * s + vv * c);
      }
    }
    // L2 norm over D=64 (4 accs x 16 lanes) + sqk scale (SQK_INIT/BASE = 32)
    float rn[8];
#pragma unroll
    for (int r = 0; r < 8; ++r) {
      float sq = 0.f;
#pragma unroll
      for (int t = 0; t < 4; ++t) sq += acc[t][r] * acc[t][r];
      sq += __shfl_xor(sq, 1, 32);
      sq += __shfl_xor(sq, 2, 32);
      sq += __shfl_xor(sq, 4, 32);
      sq += __shfl_xor(sq, 8, 32);
      rn[r] = rsqrtf(sq);
    }
    __bf16* dst = (which == 0) ? qb : kb;
#pragma unroll
    for (int t = 0; t < 4; ++t) {
      const int d = t * 16 + cb;
      const float sc = sqk[n0 + d] * 32.0f;
#pragma unroll
      for (int r = 0; r < 8; ++r)
        dst[((size_t)bh * S_ + (s0 + r)) * D_ + d] = (__bf16)(acc[t][r] * rn[r] * sc);
    }
  } else {
    // V: store transposed (B,H,D,S); rows r are contiguous in S -> b128 stores
#pragma unroll
    for (int t = 0; t < 4; ++t) {
      const int d = t * 16 + cb;
      PackU p;
#pragma unroll
      for (int r = 0; r < 8; ++r) p.h[r] = (__bf16)acc[t][r];
      *(v4u*)(vtb + ((size_t)bh * D_ + d) * S_ + s0) = p.u;
    }
  }
}

// =====================================================================
// Kernel 2: flash attention. grid (S/128, B*H), 8 waves x 16 query rows.
// K/V tiles staged via GLOBAL_LOAD_ASYNC_TO_LDS_B128 into ping-pong LDS
// buffers (ASYNCcnt-tracked DMA overlaps current tile's WMMAs; single
// barrier per iteration). Per 32-key tile: 4 WMMAs scores + online
// softmax (exp2-domain) + 4 WMMAs P*V.
// =====================================================================
__global__ __launch_bounds__(256)
void flash_attn_kernel(const __bf16* __restrict__ qb,
                       const __bf16* __restrict__ kb,
                       const __bf16* __restrict__ vtb,
                       float* __restrict__ out) {
  __shared__ __align__(16) __bf16 lK[2][32][72];    // ping-pong: 32 keys x 64 d
  __shared__ __align__(16) __bf16 lV[2][64][40];    // ping-pong: 64 d x 32 keys
  __shared__ __align__(16) __bf16 lP[8][16][40];    // per-wave P scratch 16x32

  const int tid  = threadIdx.x;
  const int wv   = tid >> 5;
  const int lane = tid & 31;
  const int hi   = lane >> 4;
  const int cb   = lane & 15;
  const int bh   = blockIdx.y;
  const int q0   = blockIdx.x * 128 + wv * 16;

  // Persistent Q A-fragments: two K-chunks of D=64
  v16bf aq[2];
  {
    const __bf16* qrow = qb + ((size_t)bh * S_ + q0 + cb) * D_;
#pragma unroll
    for (int c = 0; c < 2; ++c) {
      FragU f;
      f.u[0] = *(const v4u*)(qrow + c * 32 + hi * 8);
      f.u[1] = *(const v4u*)(qrow + c * 32 + hi * 8 + 16);
      aq[c] = f.v;
    }
  }

  const v8f vzero = {0.f, 0.f, 0.f, 0.f, 0.f, 0.f, 0.f, 0.f};
  v8f o[4];
#pragma unroll
  for (int t = 0; t < 4; ++t) o[t] = vzero;
  float mrun[8], lrun[8];
#pragma unroll
  for (int r = 0; r < 8; ++r) { mrun[r] = -1e30f; lrun[r] = 0.f; }

  // async staging setup: each thread moves 16B of K and 16B of V per tile
  const int trow = tid >> 3, tseg = tid & 7;   // K: 32 rows x 8 segs of 8 elems
  const int vrow = tid >> 2, vseg = tid & 3;   // V: 64 rows x 4 segs of 8 elems
  const void* kbase = (const void*)(kb  + (size_t)bh * S_ * D_);
  const void* vbase = (const void*)(vtb + (size_t)bh * D_ * S_);
  const unsigned kvo0 = (unsigned)((trow * D_ + tseg * 8) * sizeof(__bf16));
  const unsigned vvo0 = (unsigned)((vrow * S_ + vseg * 8) * sizeof(__bf16));
  const unsigned ldsK[2] = {(unsigned)(uintptr_t)&lK[0][trow][tseg * 8],
                            (unsigned)(uintptr_t)&lK[1][trow][tseg * 8]};
  const unsigned ldsV[2] = {(unsigned)(uintptr_t)&lV[0][vrow][vseg * 8],
                            (unsigned)(uintptr_t)&lV[1][vrow][vseg * 8]};

  // prologue: DMA tile 0 into buffer 0
  async_ld_b128(ldsK[0], kvo0, kbase);
  async_ld_b128(ldsV[0], vvo0, vbase);

  for (int kt = 0; kt < S_ / 32; ++kt) {
    const int buf = kt & 1;
    wait_async0();                      // this wave's DMA done...
    __syncthreads();                    // ...and everyone else's too

    if (kt + 1 < S_ / 32) {             // DMA next tile into the other buffer
      async_ld_b128(ldsK[buf ^ 1], kvo0 + (unsigned)((kt + 1) * 32 * D_ * 2), kbase);
      async_ld_b128(ldsV[buf ^ 1], vvo0 + (unsigned)((kt + 1) * 32 * 2), vbase);
    }

    // scores: Q(16x64) . K^T(64x32) -> two 16x16 accs (keys 0-15, 16-31)
    const v16bf bk00 = frag_b_lds(&lK[buf][0][0],   72);  // keys 0-15,  d 0-31
    const v16bf bk10 = frag_b_lds(&lK[buf][16][0],  72);  // keys 16-31, d 0-31
    const v16bf bk01 = frag_b_lds(&lK[buf][0][32],  72);  // keys 0-15,  d 32-63
    const v16bf bk11 = frag_b_lds(&lK[buf][16][32], 72);  // keys 16-31, d 32-63
    v8f sc[2];
    sc[0] = __builtin_amdgcn_wmma_f32_16x16x32_bf16(false, aq[0], false, bk00, (short)0, vzero, false, false);
    sc[1] = __builtin_amdgcn_wmma_f32_16x16x32_bf16(false, aq[0], false, bk10, (short)0, vzero, false, false);
    sc[0] = __builtin_amdgcn_wmma_f32_16x16x32_bf16(false, aq[1], false, bk01, (short)0, sc[0], false, false);
    sc[1] = __builtin_amdgcn_wmma_f32_16x16x32_bf16(false, aq[1], false, bk11, (short)0, sc[1], false, false);

    // online softmax in exp2 domain: fold D^-0.5 * log2(e) into one scale
    const float SCL = 0.125f * 1.44269504088896f;
#pragma unroll
    for (int r = 0; r < 8; ++r) {
      float s0v = sc[0][r] * SCL;
      float s1v = sc[1][r] * SCL;
      float mx = fmaxf(s0v, s1v);
      mx = fmaxf(mx, __shfl_xor(mx, 1, 32));
      mx = fmaxf(mx, __shfl_xor(mx, 2, 32));
      mx = fmaxf(mx, __shfl_xor(mx, 4, 32));
      mx = fmaxf(mx, __shfl_xor(mx, 8, 32));
      const float mnew  = fmaxf(mrun[r], mx);
      const float alpha = exp2f(mrun[r] - mnew);
      const float p0 = exp2f(s0v - mnew);
      const float p1 = exp2f(s1v - mnew);
      lrun[r] = lrun[r] * alpha + p0 + p1;   // per-lane partial row sum
      mrun[r] = mnew;
      sc[0][r] = p0; sc[1][r] = p1;
#pragma unroll
      for (int t = 0; t < 4; ++t) o[t][r] *= alpha;
    }

    // C-layout -> A-layout via per-wave LDS scratch
#pragma unroll
    for (int j = 0; j < 2; ++j)
#pragma unroll
      for (int r = 0; r < 8; ++r)
        lP[wv][r + hi * 8][j * 16 + cb] = (__bf16)sc[j][r];

    const v16bf pa  = frag_a_lds(&lP[wv][0][0], 40);
    const v16bf bv0 = frag_b_lds(&lV[buf][0][0],  40);
    const v16bf bv1 = frag_b_lds(&lV[buf][16][0], 40);
    const v16bf bv2 = frag_b_lds(&lV[buf][32][0], 40);
    const v16bf bv3 = frag_b_lds(&lV[buf][48][0], 40);
    o[0] = __builtin_amdgcn_wmma_f32_16x16x32_bf16(false, pa, false, bv0, (short)0, o[0], false, false);
    o[1] = __builtin_amdgcn_wmma_f32_16x16x32_bf16(false, pa, false, bv1, (short)0, o[1], false, false);
    o[2] = __builtin_amdgcn_wmma_f32_16x16x32_bf16(false, pa, false, bv2, (short)0, o[2], false, false);
    o[3] = __builtin_amdgcn_wmma_f32_16x16x32_bf16(false, pa, false, bv3, (short)0, o[3], false, false);
  }

  // finalize: true row sums, normalize, write (B,S,E) f32
  float inv[8];
#pragma unroll
  for (int r = 0; r < 8; ++r) {
    float l = lrun[r];
    l += __shfl_xor(l, 1, 32);
    l += __shfl_xor(l, 2, 32);
    l += __shfl_xor(l, 4, 32);
    l += __shfl_xor(l, 8, 32);
    inv[r] = 1.0f / l;
  }
  const int b = bh / H_;
  const int h = bh % H_;
#pragma unroll
  for (int t = 0; t < 4; ++t) {
    const int d = t * 16 + cb;
#pragma unroll
    for (int r = 0; r < 8; ++r) {
      const int s = q0 + hi * 8 + r;
      out[((size_t)b * S_ + s) * E_ + h * 64 + d] = o[t][r] * inv[r];
    }
  }
}

extern "C" void kernel_launch(void* const* d_in, const int* in_sizes, int n_in,
                              void* d_out, int out_size, void* d_ws, size_t ws_size,
                              hipStream_t stream) {
  (void)in_sizes; (void)n_in; (void)out_size; (void)ws_size;
  const float* x    = (const float*)d_in[0];
  const float* Wq   = (const float*)d_in[1];
  const float* Wk   = (const float*)d_in[2];
  const float* Wv   = (const float*)d_in[3];
  const float* sqk  = (const float*)d_in[4];
  const float* fcos = (const float*)d_in[5];
  const float* fsin = (const float*)d_in[6];
  float* out = (float*)d_out;

  const size_t elems = (size_t)B_ * H_ * S_ * D_;   // 8M elements
  __bf16* qb   = (__bf16*)d_ws;                     // 16 MB
  __bf16* kbuf = qb + elems;                        // 16 MB
  __bf16* vtb  = kbuf + elems;                      // 16 MB

  dim3 g1(B_ * S_ / 128, 3 * H_), blk(256);
  qkv_rope_norm_kernel<<<g1, blk, 0, stream>>>(x, Wq, Wk, Wv, sqk, fcos, fsin,
                                               qb, kbuf, vtb);
  dim3 g2(S_ / 128, B_ * H_);
  flash_attn_kernel<<<g2, blk, 0, stream>>>(qb, kbuf, vtb, out);
}
